// LocalDenseSynthesizerAttention2_3770981286602
// MI455X (gfx1250) — compile-verified, matrix operand order checked by reading, not run on hardware
//
#include <hip/hip_runtime.h>
#include <hip/hip_bf16.h>
#include <math.h>

// ---------------------------------------------------------------------------
// LocalDenseSynthesizerAttention on MI455X (gfx1250, wave32, WMMA + TDM).
// B=2 T=4096 H=4 C=63 DK=64 NF=256.  fp32 I/O; GEMMs run through
// v_wmma_f32_16x16x32_bf16; tiles staged into LDS by the Tensor Data Mover
// (tensor_load_to_lds, TENSORcnt-tracked, double-buffered).
// ---------------------------------------------------------------------------

typedef __attribute__((ext_vector_type(16))) __bf16       v16bf;
typedef __attribute__((ext_vector_type(8)))  float        v8f;
typedef __attribute__((ext_vector_type(4)))  unsigned int u32x4;
typedef __attribute__((ext_vector_type(8)))  int          i32x8;
typedef __attribute__((ext_vector_type(4)))  int          i32x4;

#define TQ    4096
#define NFEAT 256
#define HH    4
#define CC    63
#define DKK   64
#define NW    252   /* H*C */
#define HALFW 31    /* (C-1)/2 */

#if __has_builtin(__builtin_amdgcn_tensor_load_to_lds)
#define USE_TDM 1
#else
#define USE_TDM 0
#endif

#if USE_TDM
// ---------------------------------------------------------------------------
// Issue one TDM 2-D tile load: global (row-major, stride_d0 dwords) -> LDS.
// data_size = 4 B. LDS padding: 8 DWORDs after every 32 DWORDs (pad_interval=4
// => 32 DW, pad_amount=7 => 8 DW), giving an LDS row stride of 40 dwords for a
// 32-dword tile row. Reads beyond tensor_d0/d1 return zero (handles N=252).
// D# bitfields per CDNA5 ISA 8.3/8.4: group0 {count=1, lds_addr, global_addr,
// type=2}; group1 {data_size=2, pad, dims, tile dims, stride}. Groups 2/3
// (and the trailing dwords) are zero: 2-D tile, no iteration, no gather.
// 6-arg builtin form (clang-23 / therock-10.0 toolchain).
// ---------------------------------------------------------------------------
__device__ __forceinline__ void tdm_load_tile_2d(
    const float* gsrc, unsigned lds_byte_addr,
    unsigned tensor_d0, unsigned tensor_d1, unsigned stride_d0,
    unsigned tile_d0, unsigned tile_d1) {
  const unsigned long long ga = (unsigned long long)(size_t)gsrc;
  u32x4 g0;
  g0[0] = 1u;                                                 // count=1 (valid user D#)
  g0[1] = lds_byte_addr;                                      // lds_addr [63:32]
  g0[2] = (unsigned)ga;                                       // global_addr [95:64]
  g0[3] = (unsigned)((ga >> 32) & 0x01FFFFFFull) | (2u << 30);// addr[56:32] | type=2
  i32x8 g1;
  g1[0] = (int)((2u << 16)      /* data_size = 4 bytes        */
              | (1u << 20)      /* pad_enable                 */
              | (4u << 22)      /* pad_interval: every 32 DW  */
              | (7u << 25));    /* pad_amount: 8 DW           */
  g1[1] = (int)((tensor_d0 & 0xFFFFu) << 16);                 // td0[15:0] @ bits 63:48
  g1[2] = (int)(((tensor_d0 >> 16) & 0xFFFFu)                 // td0[31:16]
              | ((tensor_d1 & 0xFFFFu) << 16));               // td1[15:0]
  g1[3] = (int)(((tensor_d1 >> 16) & 0xFFFFu)                 // td1[31:16]
              | ((tile_d0 & 0xFFFFu) << 16));                 // tile_dim0
  g1[4] = (int)(tile_d1 & 0xFFFFu);                           // tile_dim1 (tile_dim2=0)
  g1[5] = (int)stride_d0;                                     // tensor_dim0_stride[31:0]
  g1[6] = 0;                                                  // stride hi / dim1_stride
  g1[7] = 0;
  const i32x4 z4 = {0, 0, 0, 0};
  const i32x8 z8 = {0, 0, 0, 0, 0, 0, 0, 0};
  __builtin_amdgcn_tensor_load_to_lds(g0, g1, z4, z4, z8, 0);
}
#endif

// ---------------------------------------------------------------------------
// Build a bf16 WMMA fragment from an fp32 LDS tile (row stride 40 dwords),
// following the CDNA5 16-bit A-matrix 16x32 layout:
//   lanes 0-15: row = lane,    K chunks {0..7, 16..23}
//   lanes16-31: row = lane-16, K chunks {8..15, 24..31}
// Two contiguous 8-float runs per lane -> ds_load_b128 pairs + v_cvt_pk_bf16.
// ---------------------------------------------------------------------------
__device__ __forceinline__ v16bf frag_f32(const float* __restrict__ base,
                                          int ld, int lane) {
  const int row = lane & 15;
  const int kh8 = (lane >> 4) << 3;          // 0 or 8
  const float* p = base + row * ld + kh8;
  v16bf f;
#pragma unroll
  for (int j = 0; j < 8; ++j) {
    const int kb = (j < 4) ? (2 * j) : (16 + 2 * (j - 4));
    f[2 * j]     = (__bf16)p[kb];
    f[2 * j + 1] = (__bf16)p[kb + 1];
  }
  return f;
}

// ---------------------------------------------------------------------------
// GEMM: Cout[M x N] = act(A[M x K] @ W[N x K]^T)
// Block tile BM=128 x BN=64, K-step 32. 256 threads = 8 waves (4x2 grid),
// each wave owns a 32x32 subtile = 2x2 v_wmma_f32_16x16x32_bf16.
// TDM double-buffers fp32 tiles into LDS; wave 0 drives the DMA pipeline.
// ---------------------------------------------------------------------------
template <bool RELU>
__global__ __launch_bounds__(256) void gemm_bf16_wmma(
    const float* __restrict__ A, const float* __restrict__ W,
    float* __restrict__ Cout, int M, int N, int K) {
  constexpr int BM = 128, BN = 64, KT = 32, LDF = KT + 8;  // 40-dword LDS rows
  __shared__ float sA[2][BM * LDF];
  __shared__ float sB[2][BN * LDF];

  const int tid  = threadIdx.x;
  const int wave = tid >> 5;
  const int lane = tid & 31;
  const int wrow = wave >> 1;               // 0..3 -> 32-row band
  const int wcol = wave & 1;                // 0..1 -> 32-col band
  const int m0   = blockIdx.x * BM;
  const int n0   = blockIdx.y * BN;

  v8f acc[2][2];
#pragma unroll
  for (int i = 0; i < 2; ++i)
#pragma unroll
    for (int j = 0; j < 2; ++j)
#pragma unroll
      for (int e = 0; e < 8; ++e) acc[i][j][e] = 0.0f;

  const int NKT = K / KT;

#if USE_TDM
  const unsigned ldsA[2] = {(unsigned)(size_t)&sA[0][0], (unsigned)(size_t)&sA[1][0]};
  const unsigned ldsB[2] = {(unsigned)(size_t)&sB[0][0], (unsigned)(size_t)&sB[1][0]};
  if (wave == 0) {  // prologue: fill buffer 0 (EXEC ignored by tensor ops)
    tdm_load_tile_2d(A + (size_t)m0 * K, ldsA[0], (unsigned)K, (unsigned)(M - m0),
                     (unsigned)K, KT, BM);
    tdm_load_tile_2d(W + (size_t)n0 * K, ldsB[0], (unsigned)K, (unsigned)(N - n0),
                     (unsigned)K, KT, BN);
  }
#endif

  for (int it = 0; it < NKT; ++it) {
#if USE_TDM
    const int rb = it & 1;
    if (wave == 0) {
      if (it + 1 < NKT) {  // prefetch next pair into the other buffer
        const int kn = (it + 1) * KT;
        tdm_load_tile_2d(A + (size_t)m0 * K + kn, ldsA[rb ^ 1],
                         (unsigned)(K - kn), (unsigned)(M - m0), (unsigned)K, KT, BM);
        tdm_load_tile_2d(W + (size_t)n0 * K + kn, ldsB[rb ^ 1],
                         (unsigned)(K - kn), (unsigned)(N - n0), (unsigned)K, KT, BN);
        __builtin_amdgcn_s_wait_tensorcnt(2);   // current pair landed
      } else {
        __builtin_amdgcn_s_wait_tensorcnt(0);   // last pair landed
      }
    }
#else
    // Fallback staging path (no TDM builtin): cooperative fp32 copy, buffer 0.
    const int rb = 0;
    const int kt = it * KT;
    if (it) __syncthreads();
    {
      const int ar = tid >> 1, ah = (tid & 1) << 4;       // A: 16 floats/thread
      const float4* ga = (const float4*)(A + (size_t)(m0 + ar) * K + kt + ah);
      float4* da = (float4*)&sA[0][ar * LDF + ah];
#pragma unroll
      for (int q = 0; q < 4; ++q) da[q] = ga[q];
    }
    {
      const int br = tid >> 2, bh = (tid & 3) << 3;       // B: 8 floats/thread
      const int gn = n0 + br;
      float4* db = (float4*)&sB[0][br * LDF + bh];
      if (gn < N) {
        const float4* gb = (const float4*)(W + (size_t)gn * K + kt + bh);
#pragma unroll
        for (int q = 0; q < 2; ++q) db[q] = gb[q];
      } else {
        const float4 z = {0.f, 0.f, 0.f, 0.f};
        db[0] = z; db[1] = z;
      }
    }
#endif
    __syncthreads();

    const float* pa = &sA[rb][0];
    const float* pb = &sB[rb][0];
    v16bf af[2], bfv[2];
    af[0]  = frag_f32(pa + (wrow * 32 + 0)  * LDF, LDF, lane);
    af[1]  = frag_f32(pa + (wrow * 32 + 16) * LDF, LDF, lane);
    bfv[0] = frag_f32(pb + (wcol * 32 + 0)  * LDF, LDF, lane);
    bfv[1] = frag_f32(pb + (wcol * 32 + 16) * LDF, LDF, lane);
#pragma unroll
    for (int i = 0; i < 2; ++i)
#pragma unroll
      for (int j = 0; j < 2; ++j)
        acc[i][j] = __builtin_amdgcn_wmma_f32_16x16x32_bf16(
            false, af[i], false, bfv[j], (short)0, acc[i][j], false, false);
    __syncthreads();
  }

  // Epilogue: C/D layout = VGPR r: M=r (lanes 0-15), M=r+8 (lanes 16-31)
  const int kh8 = (lane >> 4) << 3;
  const int nl  = lane & 15;
#pragma unroll
  for (int i = 0; i < 2; ++i)
#pragma unroll
    for (int j = 0; j < 2; ++j)
#pragma unroll
      for (int r = 0; r < 8; ++r) {
        const int gm = m0 + wrow * 32 + i * 16 + r + kh8;
        const int gn = n0 + wcol * 32 + j * 16 + nl;
        float v = acc[i][j][r];
        if (RELU) v = fmaxf(v, 0.0f);
        if (gn < N) Cout[(size_t)gm * (size_t)N + gn] = v;
      }
}

// ---------------------------------------------------------------------------
// Banded softmax + window weighted sum. One wave32 per (b,h,t).
//   p = softmax_c( valid(t,c) ? Wgt[b,t,h,c] : -inf )
//   X[b,t,h*64 + d] = sum_c p[c] * V[b, clip(t+c-31), h*64 + d]
// Lanes own two c slots for the softmax, then d = {2*lane, 2*lane+1} for the
// accumulation (DK=64 over 32 lanes). V rows hit L2 (8 MB, ~63x reuse).
// ---------------------------------------------------------------------------
__global__ __launch_bounds__(256) void attn_window(
    const float* __restrict__ Wgt, const float* __restrict__ V,
    float* __restrict__ X) {
  const int lane = threadIdx.x & 31;
  const int wave = threadIdx.x >> 5;
  const int gid  = blockIdx.x * 8 + wave;     // (b*H + h)*T + t
  const int t  = gid % TQ;
  const int bh = gid / TQ;
  const int h  = bh % HH;
  const int b  = bh / HH;

  const int c0 = lane, c1 = lane + 32;
  const int j0 = t + c0 - HALFW, j1 = t + c1 - HALFW;
  const bool v0 = (j0 >= 0) && (j0 < TQ);
  const bool v1 = (c1 < CC) && (j1 >= 0) && (j1 < TQ);

  const float* wr = Wgt + (size_t)(b * TQ + t) * NW + h * CC;
  float s0 = v0 ? wr[c0] : -INFINITY;
  float s1 = v1 ? wr[c1] : -INFINITY;

  float mx = fmaxf(s0, s1);
#pragma unroll
  for (int m = 16; m > 0; m >>= 1) mx = fmaxf(mx, __shfl_xor(mx, m, 32));
  const float e0 = v0 ? __expf(s0 - mx) : 0.0f;
  const float e1 = v1 ? __expf(s1 - mx) : 0.0f;
  float sm = e0 + e1;
#pragma unroll
  for (int m = 16; m > 0; m >>= 1) sm += __shfl_xor(sm, m, 32);
  const float inv = 1.0f / sm;
  const float p0 = e0 * inv, p1 = e1 * inv;

  float2 acc; acc.x = 0.0f; acc.y = 0.0f;
  const float* vb = V + (size_t)(b * TQ) * NFEAT + h * DKK + 2 * lane;
#pragma unroll 1
  for (int c = 0; c < CC; ++c) {
    const float pc = __shfl((c < 32) ? p0 : p1, c & 31, 32);
    int j = t + c - HALFW;
    j = (j < 0) ? 0 : ((j > TQ - 1) ? TQ - 1 : j);
    const float2 vv = *(const float2*)(vb + (size_t)j * NFEAT);
    acc.x += pc * vv.x;
    acc.y += pc * vv.y;
  }
  float* xo = X + (size_t)(b * TQ + t) * NFEAT + h * DKK + 2 * lane;
  xo[0] = acc.x;
  xo[1] = acc.y;
}

// ---------------------------------------------------------------------------
// Launch: 4 WMMA GEMMs + 1 softmax/window kernel.
// Inputs (setup order): query, key(unused), value, mask(unused), w1, w2, w3, w_out
// Workspace: X1 (8 MB) | Wgt (7.9 MB) | V (8 MB) | X (8 MB)  ~= 32 MB
// ---------------------------------------------------------------------------
extern "C" void kernel_launch(void* const* d_in, const int* in_sizes, int n_in,
                              void* d_out, int out_size, void* d_ws, size_t ws_size,
                              hipStream_t stream) {
  (void)in_sizes; (void)n_in; (void)out_size; (void)ws_size;
  const float* query = (const float*)d_in[0];
  const float* value = (const float*)d_in[2];
  const float* w1    = (const float*)d_in[4];
  const float* w2    = (const float*)d_in[5];
  const float* w3    = (const float*)d_in[6];
  const float* w_out = (const float*)d_in[7];
  float* out = (float*)d_out;

  const int M = 2 * TQ;                      // 8192 rows
  float* X1  = (float*)d_ws;                 // relu(query @ w1^T)   M x 256
  float* Wgt = X1 + (size_t)M * NFEAT;       // X1 @ w2^T            M x 252
  float* V   = Wgt + (size_t)M * NW;         // value @ w3^T         M x 256
  float* X   = V + (size_t)M * NFEAT;        // attention output     M x 256

  const dim3 blk(256);
  const dim3 g256(M / 128, NFEAT / 64);      // 64 x 4
  const dim3 g252(M / 128, (NW + 63) / 64);  // 64 x 4 (guarded)

  gemm_bf16_wmma<true ><<<g256, blk, 0, stream>>>(query, w1,   X1, M, NFEAT, NFEAT);
  gemm_bf16_wmma<false><<<g252, blk, 0, stream>>>(X1,    w2,  Wgt, M, NW,    NFEAT);
  gemm_bf16_wmma<false><<<g256, blk, 0, stream>>>(value, w3,    V, M, NFEAT, NFEAT);
  attn_window<<<dim3((2 * HH * TQ) / 8), blk, 0, stream>>>(Wgt, V, X);
  gemm_bf16_wmma<false><<<g256, blk, 0, stream>>>(X, w_out,  out, M, NFEAT, NFEAT);
}
